// SA_CLSTM_cell_5102421148227
// MI455X (gfx1250) — matrix operand order repaired and checked by reading, not compiled, not run on hardware
//
#include <hip/hip_runtime.h>
#include <hip/hip_bf16.h>

// ---------------------------------------------------------------------------
// SA-ConvLSTM cell for MI455X (gfx1250), wave32 + WMMA f32_16x16x32_f16,
// with CDNA5 async global->LDS staging in the attention kernel.
// B=8, T=8, CIN=64, FN=64, H=W=32 (HW=1024), GROUPS=8.
// ---------------------------------------------------------------------------

#define B_    8
#define T_    8
#define HW_   1024
#define FN_   64
#define STATE (B_ * FN_ * HW_)        // 524288 elems per state tensor

typedef __attribute__((ext_vector_type(16))) _Float16 v16h;
typedef __attribute__((ext_vector_type(8)))  _Float16 v8h;
typedef __attribute__((ext_vector_type(8)))  float    v8f;

// ---------------- WMMA fragment helpers (ISA 7.12.2 layouts) ---------------
// A (16x32 f16): lane L -> row M = L%16; halves e=0..7 at K = k0+(L/16)*8+e,
//                e=8..15 at K = k0+16+(L/16)*8+(e-8).  Two 16B loads.
__device__ __forceinline__ v16h frag_a(const _Float16* __restrict__ row, int k0, int lane) {
  const int h8 = (lane >> 4) << 3;
  v8h lo = *(const v8h*)(row + k0 + h8);
  v8h hi = *(const v8h*)(row + k0 + 16 + h8);
  return __builtin_shufflevector(lo, hi, 0, 1, 2, 3, 4, 5, 6, 7, 8, 9, 10, 11, 12, 13, 14, 15);
}
// B (32x16 f16): lane L -> col N = L%16; halves e=0..7 at K = k0+(L/16)*8+e,
// e=8..15 at K = k0+16+(L/16)*8+(e-8). Two 16B loads. Requires B stored [N][K].
__device__ __forceinline__ v16h frag_b(const _Float16* __restrict__ row, int k0, int lane) {
  const int off = k0 + ((lane >> 4) << 3);
  v8h lo = *(const v8h*)(row + off);
  v8h hi = *(const v8h*)(row + off + 16);
  return __builtin_shufflevector(lo, hi, 0, 1, 2, 3, 4, 5, 6, 7, 8, 9, 10, 11, 12, 13, 14, 15);
}
__device__ __forceinline__ v8f wmma32(v16h a, v16h b, v8f c) {
  return __builtin_amdgcn_wmma_f32_16x16x32_f16(false, a, false, b, (short)0, c, false, false);
}
__device__ __forceinline__ float sigf(float x) { return 1.0f / (1.0f + __expf(-x)); }

// ---------------------------------------------------------------------------
// Generic GEMM:  C[bz][m][n] = sum_k A[bz][m][k] * B[bz][n][k]  (+ bias[m])
// A: f16 [M][K] row-major.  B: f16 [N][K] row-major (i.e. operand transposed).
// Each wave computes a 16(M) x 64(N) tile: A-fragment reused across 4 WMMAs.
// ---------------------------------------------------------------------------
__global__ __launch_bounds__(256) void gemm_nt_wmma(
    const _Float16* __restrict__ A, const _Float16* __restrict__ Bm,
    const float* __restrict__ bias, float* __restrict__ C,
    int M, int N, int K, long long sA, long long sB, long long sC, int tilesTotal) {
  int wid = blockIdx.x * (blockDim.x >> 5) + (threadIdx.x >> 5);
  if (wid >= tilesTotal) return;
  const int lane = threadIdx.x & 31;
  const int ngrp = N >> 6;
  const int tpb  = (M >> 4) * ngrp;
  const int bz = wid / tpb;
  const int t  = wid - bz * tpb;
  const int mt = t / ngrp;
  const int ng = t - mt * ngrp;
  const int m0 = mt << 4, n0 = ng << 6;

  const _Float16* arow = A  + (size_t)sA * bz + (size_t)(m0 + (lane & 15)) * K;
  const _Float16* brow = Bm + (size_t)sB * bz + (size_t)(n0 + (lane & 15)) * K;
  v8f acc0 = {}, acc1 = {}, acc2 = {}, acc3 = {};
  for (int k0 = 0; k0 < K; k0 += 32) {
    v16h a  = frag_a(arow, k0, lane);
    v16h b0 = frag_b(brow,              k0, lane);
    v16h b1 = frag_b(brow + 16 * K,     k0, lane);
    v16h b2 = frag_b(brow + 32 * K,     k0, lane);
    v16h b3 = frag_b(brow + 48 * K,     k0, lane);
    acc0 = wmma32(a, b0, acc0);
    acc1 = wmma32(a, b1, acc1);
    acc2 = wmma32(a, b2, acc2);
    acc3 = wmma32(a, b3, acc3);
  }
  const int nl = lane & 15, hi = lane >> 4;
  float* cbase = C + (size_t)sC * bz;
  for (int v = 0; v < 8; ++v) {
    int m = m0 + v + (hi << 3);
    float bv = bias ? bias[m] : 0.0f;
    size_t r = (size_t)m * N;
    cbase[r + n0 +  0 + nl] = acc0[v] + bv;
    cbase[r + n0 + 16 + nl] = acc1[v] + bv;
    cbase[r + n0 + 32 + nl] = acc2[v] + bv;
    cbase[r + n0 + 48 + nl] = acc3[v] + bv;
  }
}

// ---------------------------------------------------------------------------
// 3x3 SAME conv as implicit GEMM: gates[b][o][pix] over 9 shifted taps.
// Wh: f16 [9][256][128] (tap-major, weights row-major over ci).
// Ch: f16 [B][1024pix][128ch] channel-last (B-operand layout).
// Each wave: 64(M out-ch) x 16(N pix) tile; B-fragment reused across 4 WMMAs.
// ---------------------------------------------------------------------------
__global__ __launch_bounds__(256) void conv3x3_wmma(
    const _Float16* __restrict__ Wh, const _Float16* __restrict__ Ch,
    const float* __restrict__ cb, float* __restrict__ gates) {
  int wid = blockIdx.x * (blockDim.x >> 5) + (threadIdx.x >> 5);
  if (wid >= B_ * 4 * 64) return;
  const int lane = threadIdx.x & 31;
  const int b  = wid >> 8;
  const int r  = wid & 255;
  const int mg = r >> 6;           // 0..3  (group of 64 out channels)
  const int nt = r & 63;           // pixel tile (16 pixels, half an image row)
  const int pix = (nt << 4) + (lane & 15);
  const int y = pix >> 5, x = pix & 31;

  v8f acc[4] = {{}, {}, {}, {}};
  for (int tap = 0; tap < 9; ++tap) {
    const int dy = tap / 3 - 1, dx = tap % 3 - 1;
    const int yy = y + dy, xx = x + dx;
    const bool ok = (yy >= 0) && (yy < 32) && (xx >= 0) && (xx < 32);
    const _Float16* brow =
        Ch + (((size_t)b * 1024 + (size_t)(ok ? (yy * 32 + xx) : 0)) << 7);
    const _Float16* wb = Wh + (size_t)tap * 256 * 128;
    for (int k0 = 0; k0 < 128; k0 += 32) {
      v16h bf = {};
      if (ok) bf = frag_b(brow, k0, lane);
      for (int mi = 0; mi < 4; ++mi) {
        int m0 = (mg << 6) + (mi << 4);
        v16h af = frag_a(wb + (size_t)(m0 + (lane & 15)) * 128, k0, lane);
        acc[mi] = wmma32(af, bf, acc[mi]);
      }
    }
  }
  const int hi = lane >> 4;
  for (int mi = 0; mi < 4; ++mi)
    for (int v = 0; v < 8; ++v) {
      int m = (mg << 6) + (mi << 4) + v + (hi << 3);
      gates[(((size_t)b * 256 + m) << 10) + pix] = acc[mi][v] + cb[m];
    }
}

// ---------------------------------------------------------------------------
// Attention logits + softmax (softmax over m, per query-row n).
// Qt,Kt: f16 [B][1024][64] channel-last.  Output Pt: f16 [B][m][n] (transposed
// probabilities, ready as the B-operand of the PV GEMM).
// One WG per (b, 16-row strip): 16x1024 f32 logits in LDS (exactly 64 KB).
// The shared 16x64 Q strip (2 KB) is staged once per WG into LDS with the
// CDNA5 async copy path (global_load_async_to_lds_b128 + s_wait_asynccnt),
// aliasing the head of the logit buffer; fragments are consumed into VGPRs
// (ds_load_b128) before logits overwrite the region.
// ---------------------------------------------------------------------------
__global__ __launch_bounds__(256) void attn_wmma(
    const _Float16* __restrict__ Qt, const _Float16* __restrict__ Kt,
    _Float16* __restrict__ Pt) {
  __shared__ float S[16 * 1024];
  const int b = blockIdx.y;
  const int n0 = blockIdx.x << 4;
  const int lane = threadIdx.x & 31;
  const int wave = threadIdx.x >> 5;
  const int tid  = threadIdx.x;

  // ---- stage Q strip (contiguous 2048 B) into LDS head, async ----
  _Float16* Sh = (_Float16*)S;                       // 1024 halves = 2 KB
  const _Float16* qbase = Qt + ((size_t)b * 1024 + n0) * 64;
#if defined(__HIP_DEVICE_COMPILE__)
  if (tid < 128) {
    unsigned ldsOff = (unsigned)(uintptr_t)(Sh + tid * 8);          // 16 B/lane
    unsigned long long ga = (unsigned long long)(uintptr_t)(qbase + tid * 8);
    asm volatile("global_load_async_to_lds_b128 %0, %1, off"
                 :: "v"(ldsOff), "v"(ga) : "memory");
  }
#if defined(__has_builtin) && __has_builtin(__builtin_amdgcn_s_wait_asynccnt)
  __builtin_amdgcn_s_wait_asynccnt(0);
#else
  asm volatile("s_wait_asynccnt 0x0" ::: "memory");
#endif
#else
  if (tid < 128) {
    ((ulonglong2*)Sh)[tid] = ((const ulonglong2*)qbase)[tid];
  }
#endif
  __syncthreads();

  // A fragments from LDS (Q strip is [16 rows][64 halves] row-major).
  const _Float16* qrow = Sh + (size_t)(lane & 15) * 64;
  v16h a0 = frag_a(qrow, 0, lane);
  v16h a1 = frag_a(qrow, 32, lane);
  __syncthreads();   // all waves hold Q in VGPRs before S is reused for logits

  for (int mt = wave; mt < 64; mt += 8) {
    const int m0 = mt << 4;
    const _Float16* krow = Kt + ((size_t)b * 1024 + m0 + (lane & 15)) * 64;
    v16h b0 = frag_b(krow, 0, lane);
    v16h b1 = frag_b(krow, 32, lane);
    v8f c = {};
    c = wmma32(a0, b0, c);
    c = wmma32(a1, b1, c);
    const int nl = lane & 15, hi = lane >> 4;
    for (int v = 0; v < 8; ++v)
      S[((v + (hi << 3)) << 10) + m0 + nl] = c[v];
  }
  __syncthreads();

  const int row = threadIdx.x >> 4;   // 0..15
  const int t   = threadIdx.x & 15;
  float mx = -3.0e38f;
  for (int j = t; j < 1024; j += 16) mx = fmaxf(mx, S[(row << 10) + j]);
  for (int o = 8; o; o >>= 1) mx = fmaxf(mx, __shfl_xor(mx, o, 16));
  float sm = 0.0f;
  for (int j = t; j < 1024; j += 16) sm += __expf(S[(row << 10) + j] - mx);
  for (int o = 8; o; o >>= 1) sm += __shfl_xor(sm, o, 16);
  const float rinv = 1.0f / sm;

  _Float16* pb = Pt + ((size_t)b << 20);
  for (int j = t; j < 1024; j += 16) {
    float p = __expf(S[(row << 10) + j] - mx) * rinv;
    pb[((size_t)j << 10) + n0 + row] = (_Float16)p;   // transposed store
  }
}

// ----------------------------- pointwise kernels ---------------------------
__global__ void zero_f32(float* p, int n) {
  int i = blockIdx.x * 256 + threadIdx.x;
  if (i < n) p[i] = 0.0f;
}
__global__ void cvt_f16_k(const float* __restrict__ s, _Float16* __restrict__ d, int n) {
  int i = blockIdx.x * 256 + threadIdx.x;
  if (i < n) d[i] = (_Float16)s[i];
}
// conv_w [256][128][3][3] -> Wh [9][256][128]
__global__ void cvt_convw_k(const float* __restrict__ w, _Float16* __restrict__ Wh) {
  int i = blockIdx.x * 256 + threadIdx.x;
  if (i >= 9 * 256 * 128) return;
  int tap = i >> 15, rem = i & 32767;
  int o = rem >> 7, ci = rem & 127;
  int ky = tap / 3, kx = tap % 3;
  Wh[i] = (_Float16)w[((size_t)(o * 128 + ci)) * 9 + ky * 3 + kx];
}
// comb_h [b][pix][c] = concat(x_t, ih) channel-last f16
__global__ void pack_comb_k(const float* __restrict__ xin, const float* __restrict__ ih,
                            _Float16* __restrict__ Ch, int t) {
  int i = blockIdx.x * 256 + threadIdx.x;
  if (i >= B_ * 1024 * 128) return;
  int b = i >> 17, pix = (i >> 7) & 1023, c = i & 127;
  float v = (c < 64)
      ? xin[((((size_t)b * T_ + t) * 64 + c) << 10) + pix]
      : ih[(((size_t)b * 64 + (c - 64)) << 10) + pix];
  Ch[i] = (_Float16)v;
}
// GroupNorm stats per (b, group): gates[b][256][1024], group = 32 channels.
__global__ void gn_stats_k(const float* __restrict__ gates, float* __restrict__ stats) {
  __shared__ float s1[256], s2[256];
  int bg = blockIdx.x;
  const float* base = gates + ((size_t)bg << 15);   // 32*1024 per group
  float a = 0.0f, q = 0.0f;
  for (int i = threadIdx.x; i < 32768; i += 256) {
    float v = base[i]; a += v; q += v * v;
  }
  s1[threadIdx.x] = a; s2[threadIdx.x] = q;
  __syncthreads();
  for (int o = 128; o; o >>= 1) {
    if (threadIdx.x < (unsigned)o) {
      s1[threadIdx.x] += s1[threadIdx.x + o];
      s2[threadIdx.x] += s2[threadIdx.x + o];
    }
    __syncthreads();
  }
  if (threadIdx.x == 0) {
    float mean = s1[0] * (1.0f / 32768.0f);
    float var  = s2[0] * (1.0f / 32768.0f) - mean * mean;
    stats[bg * 2]     = mean;
    stats[bg * 2 + 1] = rsqrtf(var + 1e-5f);
  }
}
// GN-apply + LSTM gates; also packs f16 operands for the 1x1 GEMMs.
__global__ void lstm_pw_k(const float* __restrict__ gates, const float* __restrict__ stats,
                          const float* __restrict__ gg, const float* __restrict__ gb,
                          float* __restrict__ ic, const float* __restrict__ im,
                          float* __restrict__ ohb, _Float16* __restrict__ oh_h,
                          _Float16* __restrict__ im_h) {
  int i = blockIdx.x * 256 + threadIdx.x;
  if (i >= STATE) return;
  int b = i >> 16, c = (i >> 10) & 63, pix = i & 1023;
  float xn[4];
  for (int j = 0; j < 4; ++j) {
    int ch = j * 64 + c;
    int g = ch >> 5;
    float mean = stats[(b * 8 + g) * 2];
    float rstd = stats[(b * 8 + g) * 2 + 1];
    float v = gates[(((size_t)b * 256 + ch) << 10) + pix];
    xn[j] = (v - mean) * rstd * gg[ch] + gb[ch];
  }
  float ig = sigf(xn[0]), fg = sigf(xn[1]), cg = tanhf(xn[2]), og = sigf(xn[3]);
  float oc = ic[i] * fg + ig * cg;
  float oh = og * tanhf(oc);
  ic[i] = oc;
  ohb[i] = oh;
  size_t cl = ((size_t)b << 16) + ((size_t)pix << 6) + c;
  oh_h[cl] = (_Float16)oh;
  im_h[cl] = (_Float16)im[i];
}
// qkv[b][192][1024], mkv[b][128][1024] -> attention operand layouts.
__global__ void pack_qkv_k(const float* __restrict__ qkv, const float* __restrict__ mkv,
                           _Float16* __restrict__ Qt, _Float16* __restrict__ Kt,
                           _Float16* __restrict__ Kmt, _Float16* __restrict__ Vh,
                           _Float16* __restrict__ Vm) {
  int i = blockIdx.x * 256 + threadIdx.x;
  if (i >= STATE) return;
  int b = i >> 16, c = (i >> 10) & 63, n = i & 1023;
  size_t qb = ((size_t)b * 192) << 10;
  size_t mb = ((size_t)b * 128) << 10;
  size_t cl = ((size_t)b << 16) + ((size_t)n << 6) + c;   // channel-last
  size_t cf = ((size_t)b << 16) + ((size_t)c << 10) + n;  // channel-first
  Qt[cl]  = (_Float16)qkv[qb + ((size_t)c << 10) + n];
  Kt[cl]  = (_Float16)qkv[qb + ((size_t)(64 + c) << 10) + n];
  Vh[cf]  = (_Float16)qkv[qb + ((size_t)(128 + c) << 10) + n];
  Kmt[cl] = (_Float16)mkv[mb + ((size_t)c << 10) + n];
  Vm[cf]  = (_Float16)mkv[mb + ((size_t)(64 + c) << 10) + n];
}
__global__ void pack_zcat_k(const float* __restrict__ Zh, const float* __restrict__ Zm,
                            _Float16* __restrict__ Zc) {
  int i = blockIdx.x * 256 + threadIdx.x;
  if (i >= B_ * 1024 * 128) return;
  int b = i >> 17, pix = (i >> 7) & 1023, c = i & 127;
  float v = (c < 64) ? Zh[((size_t)b << 16) + ((size_t)c << 10) + pix]
                     : Zm[((size_t)b << 16) + ((size_t)(c - 64) << 10) + pix];
  Zc[i] = (_Float16)v;
}
__global__ void pack_cat2_k(const float* __restrict__ Z, const float* __restrict__ ohb,
                            _Float16* __restrict__ C2) {
  int i = blockIdx.x * 256 + threadIdx.x;
  if (i >= B_ * 1024 * 128) return;
  int b = i >> 17, pix = (i >> 7) & 1023, c = i & 127;
  float v = (c < 64) ? Z[((size_t)b << 16) + ((size_t)c << 10) + pix]
                     : ohb[((size_t)b << 16) + ((size_t)(c - 64) << 10) + pix];
  C2[i] = (_Float16)v;
}
__global__ void sam_pw_k(const float* __restrict__ g2, float* __restrict__ im,
                         float* __restrict__ ih, float* __restrict__ out, int t) {
  int i = blockIdx.x * 256 + threadIdx.x;
  if (i >= STATE) return;
  int b = i >> 16, c = (i >> 10) & 63, pix = i & 1023;
  size_t gb = ((size_t)b * 192) << 10;
  float o  = g2[gb + ((size_t)c << 10) + pix];
  float g  = g2[gb + ((size_t)(64 + c) << 10) + pix];
  float ii = sigf(g2[gb + ((size_t)(128 + c) << 10) + pix]);
  float om  = tanhf(g) * ii + (1.0f - ii) * im[i];
  float oh2 = sigf(o) * om;
  im[i] = om;
  ih[i] = oh2;
  out[((((size_t)b * T_ + t) * 64 + c) << 10) + pix] = oh2;
}
__global__ void copy_states_k(const float* __restrict__ ih, const float* __restrict__ ic,
                              const float* __restrict__ im, float* __restrict__ out) {
  int i = blockIdx.x * 256 + threadIdx.x;
  if (i >= 3 * STATE) return;
  int sel = i / STATE, r = i - sel * STATE;
  const float* s = (sel == 0) ? ih : ((sel == 1) ? ic : im);
  out[(size_t)B_ * T_ * FN_ * HW_ + i] = s[r];
}

// ---------------------------------------------------------------------------
static inline void launch_gemm(const _Float16* A, const _Float16* B, const float* bias,
                               float* C, int M, int N, int K,
                               long long sA, long long sB, long long sC,
                               int batches, hipStream_t st) {
  int tiles = batches * (M / 16) * (N / 64);
  int blocks = (tiles + 7) / 8;
  gemm_nt_wmma<<<blocks, 256, 0, st>>>(A, B, bias, C, M, N, K, sA, sB, sC, tiles);
}

extern "C" void kernel_launch(void* const* d_in, const int* in_sizes, int n_in,
                              void* d_out, int out_size, void* d_ws, size_t ws_size,
                              hipStream_t stream) {
  (void)in_sizes; (void)n_in; (void)out_size; (void)ws_size;
  const float* inputs = (const float*)d_in[0];
  const float* conv_w = (const float*)d_in[1];
  const float* conv_b = (const float*)d_in[2];
  const float* gn_g   = (const float*)d_in[3];
  const float* gn_b   = (const float*)d_in[4];
  const float* h_w    = (const float*)d_in[5];
  const float* h_b    = (const float*)d_in[6];
  const float* m_w    = (const float*)d_in[7];
  const float* m_b    = (const float*)d_in[8];
  const float* z1_w   = (const float*)d_in[9];
  const float* z1_b   = (const float*)d_in[10];
  const float* z2_w   = (const float*)d_in[11];
  const float* z2_b   = (const float*)d_in[12];
  float* out = (float*)d_out;

  // ---------------- workspace layout ----------------
  char* w = (char*)d_ws;
  size_t off = 0;
  auto nxt = [&](size_t bytes) -> void* {
    void* p = w + off;
    off += (bytes + 255) & ~(size_t)255;
    return p;
  };
  float*    ih    = (float*)   nxt((size_t)STATE * 4);
  float*    ic    = (float*)   nxt((size_t)STATE * 4);
  float*    im    = (float*)   nxt((size_t)STATE * 4);
  _Float16* Wh    = (_Float16*)nxt((size_t)9 * 256 * 128 * 2);
  _Float16* hWh   = (_Float16*)nxt((size_t)192 * 64 * 2);
  _Float16* mWh   = (_Float16*)nxt((size_t)128 * 64 * 2);
  _Float16* z1Wh  = (_Float16*)nxt((size_t)64 * 128 * 2);
  _Float16* z2Wh  = (_Float16*)nxt((size_t)192 * 128 * 2);
  _Float16* Ch    = (_Float16*)nxt((size_t)B_ * 1024 * 128 * 2);
  float*    gates = (float*)   nxt((size_t)B_ * 256 * 1024 * 4);
  float*    stats = (float*)   nxt((size_t)128 * 4);
  float*    ohb   = (float*)   nxt((size_t)STATE * 4);
  _Float16* oh_h  = (_Float16*)nxt((size_t)STATE * 2);
  _Float16* im_h  = (_Float16*)nxt((size_t)STATE * 2);
  float*    qkv   = (float*)   nxt((size_t)B_ * 192 * 1024 * 4);
  float*    mkv   = (float*)   nxt((size_t)B_ * 128 * 1024 * 4);
  _Float16* Qt    = (_Float16*)nxt((size_t)STATE * 2);
  _Float16* Kt    = (_Float16*)nxt((size_t)STATE * 2);
  _Float16* Kmt   = (_Float16*)nxt((size_t)STATE * 2);
  _Float16* Vh    = (_Float16*)nxt((size_t)STATE * 2);
  _Float16* Vm    = (_Float16*)nxt((size_t)STATE * 2);
  _Float16* Pt    = (_Float16*)nxt((size_t)B_ * 1024 * 1024 * 2);
  float*    Zh    = (float*)   nxt((size_t)STATE * 4);
  float*    Zm    = (float*)   nxt((size_t)STATE * 4);
  _Float16* Zc    = (_Float16*)nxt((size_t)B_ * 1024 * 128 * 2);
  float*    Zbuf  = (float*)   nxt((size_t)STATE * 4);
  _Float16* C2    = (_Float16*)nxt((size_t)B_ * 1024 * 128 * 2);
  float*    g2    = (float*)   nxt((size_t)B_ * 192 * 1024 * 4);

  const int TB = 256;
  auto blk = [](int n) { return (n + 255) / 256; };

  // ---------------- one-time setup (per call: deterministic) ----------------
  zero_f32<<<blk(STATE), TB, 0, stream>>>(ih, STATE);
  zero_f32<<<blk(STATE), TB, 0, stream>>>(ic, STATE);
  zero_f32<<<blk(STATE), TB, 0, stream>>>(im, STATE);
  cvt_convw_k<<<blk(9 * 256 * 128), TB, 0, stream>>>(conv_w, Wh);
  cvt_f16_k<<<blk(192 * 64), TB, 0, stream>>>(h_w, hWh, 192 * 64);
  cvt_f16_k<<<blk(128 * 64), TB, 0, stream>>>(m_w, mWh, 128 * 64);
  cvt_f16_k<<<blk(64 * 128), TB, 0, stream>>>(z1_w, z1Wh, 64 * 128);
  cvt_f16_k<<<blk(192 * 128), TB, 0, stream>>>(z2_w, z2Wh, 192 * 128);

  // ---------------- time loop ----------------
  for (int t = 0; t < T_; ++t) {
    pack_comb_k<<<blk(B_ * 1024 * 128), TB, 0, stream>>>(inputs, ih, Ch, t);
    conv3x3_wmma<<<(B_ * 4 * 64) / 8, TB, 0, stream>>>(Wh, Ch, conv_b, gates);
    gn_stats_k<<<B_ * 8, TB, 0, stream>>>(gates, stats);
    lstm_pw_k<<<blk(STATE), TB, 0, stream>>>(gates, stats, gn_g, gn_b,
                                             ic, im, ohb, oh_h, im_h);
    // h_qkv = h_w @ oh ; m_kv = m_w @ im
    launch_gemm(hWh, oh_h, h_b, qkv, 192, 1024, 64,
                0, (long long)HW_ * 64, (long long)192 * 1024, B_, stream);
    launch_gemm(mWh, im_h, m_b, mkv, 128, 1024, 64,
                0, (long long)HW_ * 64, (long long)128 * 1024, B_, stream);
    pack_qkv_k<<<blk(STATE), TB, 0, stream>>>(qkv, mkv, Qt, Kt, Kmt, Vh, Vm);
    // attention (h), then Zh = Vh @ Ah
    attn_wmma<<<dim3(64, B_), TB, 0, stream>>>(Qt, Kt, Pt);
    launch_gemm(Vh, Pt, nullptr, Zh, 64, 1024, 1024,
                (long long)64 * 1024, (long long)1024 * 1024,
                (long long)64 * 1024, B_, stream);
    // attention (m), then Zm = Vm @ Am  (Pt buffer reused)
    attn_wmma<<<dim3(64, B_), TB, 0, stream>>>(Qt, Kmt, Pt);
    launch_gemm(Vm, Pt, nullptr, Zm, 64, 1024, 1024,
                (long long)64 * 1024, (long long)1024 * 1024,
                (long long)64 * 1024, B_, stream);
    pack_zcat_k<<<blk(B_ * 1024 * 128), TB, 0, stream>>>(Zh, Zm, Zc);
    launch_gemm(z1Wh, Zc, z1_b, Zbuf, 64, 1024, 128,
                0, (long long)HW_ * 128, (long long)64 * 1024, B_, stream);
    pack_cat2_k<<<blk(B_ * 1024 * 128), TB, 0, stream>>>(Zbuf, ohb, C2);
    launch_gemm(z2Wh, C2, z2_b, g2, 192, 1024, 128,
                0, (long long)HW_ * 128, (long long)192 * 1024, B_, stream);
    sam_pw_k<<<blk(STATE), TB, 0, stream>>>(g2, im, ih, out, t);
  }
  copy_states_k<<<blk(3 * STATE), TB, 0, stream>>>(ih, ic, im, out);
}
